// BertSelfAttention_26087631356246
// MI455X (gfx1250) — compile-verified, matrix operand order checked by reading, not run on hardware
//
#include <hip/hip_runtime.h>
#include <hip/hip_bf16.h>
#include <math.h>
#include <stdint.h>

// ---------------------------------------------------------------------------
// BertSelfAttention with previous-attention head mixing, for MI455X (gfx1250).
//
// Roofline: previous_attention is 256 MB. Head-mix is materialized once, in
// f16 with the 0.5 blend factor and bias folded in: traffic = 256 MB (prev R)
// + 128 MB (mixed W) + 128 MB (mixed R) = 512 MB ~= 22 us at 23.3 TB/s.
// All matmuls run on v_wmma_f32_16x16x32_f16 / v_wmma_f32_16x16x4_f32.
// The mixed panel is pulled into LDS with global_load_async_to_lds_b128
// overlapped with the score WMMA pass; V is prefetched during softmax.
// ---------------------------------------------------------------------------

#define BB   4
#define SS   1024
#define DD   1024
#define HH   16
#define HD   64
#define ALPHA 0.5f

typedef __attribute__((ext_vector_type(16))) _Float16 v16h;
typedef __attribute__((ext_vector_type(8)))  float    v8f;
typedef __attribute__((ext_vector_type(2)))  float    v2f;

// 16-bit A/B matrix (16 x K32) lane/half -> K mapping per CDNA5 ISA 7.12.2.
__device__ __forceinline__ int a16_k(int lane, int e) {
  int vg = e >> 1;
  int lo = e & 1;
  int kb = (lane < 16) ? 0 : 8;
  return (vg < 4) ? (kb + 2 * vg + lo) : (16 + kb + 2 * (vg - 4) + lo);
}

// async global -> LDS copy of 16 bytes (ASYNCcnt-tracked, CDNA5)
__device__ __forceinline__ void async_to_lds_b128(unsigned lds_off,
                                                  const void* gaddr) {
  asm volatile("global_load_async_to_lds_b128 %0, %1, off"
               :: "v"(lds_off), "v"(gaddr) : "memory");
}
__device__ __forceinline__ void wait_asynccnt0() {
  asm volatile("s_wait_asynccnt 0x0" ::: "memory");
}

// ---------------------------------------------------------------------------
// Kernel 1: QKV projection.  Y = X @ W^T + bias, emitted as f16 in [B,H,S,HD].
// 8 waves; block tile M=32 x N=64; 32 v_wmma_f32_16x16x32_f16 per wave.
// ---------------------------------------------------------------------------
__global__ __launch_bounds__(256)
void qkv_kernel(const float* __restrict__ X,
                const float* __restrict__ Wq, const float* __restrict__ bq,
                const float* __restrict__ Wk, const float* __restrict__ bk,
                const float* __restrict__ Wv, const float* __restrict__ bv,
                _Float16* __restrict__ qf, _Float16* __restrict__ kf,
                _Float16* __restrict__ vf) {
  const float* W;  const float* bias;  _Float16* out;
  if (blockIdx.z == 0)      { W = Wq; bias = bq; out = qf; }
  else if (blockIdx.z == 1) { W = Wk; bias = bk; out = kf; }
  else                      { W = Wv; bias = bv; out = vf; }

  const int m0   = blockIdx.x * 32;
  const int n0   = blockIdx.y * 64;
  const int tid  = threadIdx.x;
  const int lane = tid & 31;
  const int wv   = tid >> 5;
  const int mt   = wv >> 2;   // 0..1
  const int nt   = wv & 3;    // 0..3

  __shared__ _Float16 Xs[32 * 32];
  __shared__ _Float16 Ws[64 * 32];

  v8f acc = {};
  for (int k0 = 0; k0 < DD; k0 += 32) {
#pragma unroll
    for (int i = 0; i < 4; ++i) {            // 1024 halves of X tile
      int e  = tid * 4 + i;
      int lr = e >> 5, lc = e & 31;
      Xs[e] = (_Float16)X[(size_t)(m0 + lr) * DD + k0 + lc];
    }
#pragma unroll
    for (int i = 0; i < 8; ++i) {            // 2048 halves of W tile
      int e  = tid * 8 + i;
      int lr = e >> 5, lc = e & 31;
      Ws[e] = (_Float16)W[(size_t)(n0 + lr) * DD + k0 + lc];
    }
    __syncthreads();

    v16h af, bf;
#pragma unroll
    for (int e = 0; e < 16; ++e) {
      int k = a16_k(lane, e);
      af[e] = Xs[(mt * 16 + (lane & 15)) * 32 + k];
      bf[e] = Ws[(nt * 16 + (lane & 15)) * 32 + k];
    }
    acc = __builtin_amdgcn_wmma_f32_16x16x32_f16(false, af, false, bf,
                                                 (short)0, acc, false, false);
    __syncthreads();
  }

  const int ncol = n0 + nt * 16 + (lane & 15);
  const float bn = bias[ncol];
  const int mb   = m0 + mt * 16 + ((lane >> 4) << 3);
  const int h    = ncol >> 6;
  const int d    = ncol & 63;
#pragma unroll
  for (int r = 0; r < 8; ++r) {
    int m = mb + r;
    int b = m >> 10, s = m & 1023;
    out[(((size_t)(b * HH + h)) * SS + s) * HD + d] = (_Float16)(acc[r] + bn);
  }
}

// ---------------------------------------------------------------------------
// Kernel 2: head mixing.  mixed[b,g,s,t] = 0.5*(sum_h Wc[g,h]*prev + bc[g])
// stored as f16 (blend factor + bias folded; halves downstream traffic).
// 4x v_wmma_f32_16x16x4_f32 per wave-tile, bias in the C accumulator.
// ---------------------------------------------------------------------------
__global__ __launch_bounds__(256)
void headmix_kernel(const float* __restrict__ prev,
                    const float* __restrict__ Wc,
                    const float* __restrict__ bc,
                    _Float16* __restrict__ mixed) {
  const int b    = blockIdx.z;
  const int s    = blockIdx.y;
  const int tid  = threadIdx.x;
  const int lane = tid & 31;
  const int wv   = tid >> 5;
  const int t0   = blockIdx.x * 128 + wv * 16;
  const int n    = lane & 15;
  const int kb   = (lane < 16) ? 0 : 2;   // f32 A/B K base per half-wave
  const int grow = lane & 15;
  const int mb   = (lane >> 4) << 3;

  v2f a[4];
#pragma unroll
  for (int kk = 0; kk < 4; ++kk) {
    int h0 = kk * 4 + kb;
    a[kk][0] = Wc[grow * 16 + h0];
    a[kk][1] = Wc[grow * 16 + h0 + 1];
  }

  v8f acc;
#pragma unroll
  for (int r = 0; r < 8; ++r) acc[r] = bc[mb + r];

  const size_t base = (size_t)b * HH * SS * SS + (size_t)s * SS;
#pragma unroll
  for (int kk = 0; kk < 4; ++kk) {
    int h0 = kk * 4 + kb;
    v2f bf;
    bf[0] = prev[base + (size_t)h0 * SS * SS + t0 + n];
    bf[1] = prev[base + (size_t)(h0 + 1) * SS * SS + t0 + n];
    acc = __builtin_amdgcn_wmma_f32_16x16x4_f32(false, a[kk], false, bf,
                                                (short)0, acc, false, false);
  }

  _Float16* mo = mixed + base;
#pragma unroll
  for (int r = 0; r < 8; ++r)
    mo[(size_t)(mb + r) * SS * SS + t0 + n] = (_Float16)(ALPHA * acc[r]);
}

// ---------------------------------------------------------------------------
// Kernel 3: fused attention for one (b, h, 16-row s-tile).
//  - async-copy mixed panel (32 KB, contiguous) into LDS, hidden behind
//  - scores panel 16x1024 via WMMA into LDS (f32, 64 KB)
//  - softmax #1, blend with LDS mixed panel, softmax #2, probs -> f16 LDS
//  - ctx = probs @ v via WMMA (V prefetched during softmax), LDS reduction
// ---------------------------------------------------------------------------
__global__ __launch_bounds__(256)
void attn_kernel(const _Float16* __restrict__ qf,
                 const _Float16* __restrict__ kf,
                 const _Float16* __restrict__ vf,
                 const _Float16* __restrict__ mixed,
                 const float* __restrict__ mask,
                 float* __restrict__ out) {
  const int s0   = blockIdx.x * 16;
  const int h    = blockIdx.y;
  const int b    = blockIdx.z;
  const int tid  = threadIdx.x;
  const int lane = tid & 31;
  const int wv   = tid >> 5;

  __shared__ float     sc[16 * 1024];     // 64 KB score/work panel
  __shared__ _Float16  pf[16 * 1024];     // 32 KB final probs (f16)
  __shared__ _Float16  mixs[16 * 1024];   // 32 KB mixed panel (async-filled)
  __shared__ float     red[16 * 32];      // row reduction scratch

  const _Float16* qh = qf + (size_t)(b * HH + h) * SS * HD;
  const _Float16* kh = kf + (size_t)(b * HH + h) * SS * HD;
  const _Float16* vh = vf + (size_t)(b * HH + h) * SS * HD;

  // ---- kick off async LDS fill of the mixed panel (contiguous 32 KB) ----
  {
    const _Float16* mp = mixed + ((size_t)(b * HH + h) * SS + s0) * SS;
    unsigned lbase = (unsigned)(uintptr_t)(&mixs[0]);
#pragma unroll
    for (int c = 0; c < 8; ++c) {
      int off = (tid + c * 256) * 8;            // halves (16B chunks)
      async_to_lds_b128(lbase + (unsigned)(off * 2), (const void*)(mp + off));
    }
  }

  const int mrowA = lane & 15;

  // Q A-fragments for the two K=32 steps covering HD=64
  v16h aq0, aq1;
#pragma unroll
  for (int e = 0; e < 16; ++e) {
    int k = a16_k(lane, e);
    aq0[e] = qh[(size_t)(s0 + mrowA) * HD + k];
    aq1[e] = qh[(size_t)(s0 + mrowA) * HD + 32 + k];
  }

  // ---- scores: 64 column tiles split round-robin over 8 waves ----
  for (int ct = wv; ct < 64; ct += 8) {
    int t0 = ct * 16;
    v16h b0, b1;
    int nc = lane & 15;
#pragma unroll
    for (int e = 0; e < 16; ++e) {
      int k = a16_k(lane, e);
      b0[e] = kh[(size_t)(t0 + nc) * HD + k];
      b1[e] = kh[(size_t)(t0 + nc) * HD + 32 + k];
    }
    v8f acc = {};
    acc = __builtin_amdgcn_wmma_f32_16x16x32_f16(false, aq0, false, b0,
                                                 (short)0, acc, false, false);
    acc = __builtin_amdgcn_wmma_f32_16x16x32_f16(false, aq1, false, b1,
                                                 (short)0, acc, false, false);
    float mval = mask[(size_t)b * SS + t0 + nc];
    int mbr = (lane >> 4) << 3;
#pragma unroll
    for (int r = 0; r < 8; ++r)
      sc[(mbr + r) * 1024 + t0 + nc] = acc[r] * 0.125f + mval;  // 1/sqrt(64)
  }

  // prefetch V panel (128 KB) into cache for the ctx pass
  {
    const char* vb = (const char*)vh;
#pragma unroll
    for (int i = 0; i < 4; ++i)
      __builtin_prefetch(vb + tid * 512 + i * 128, 0, 3);
  }

  wait_asynccnt0();          // mixed panel resident in LDS
  __syncthreads();

  // ---- softmax #1, blend, softmax #2 : 16 threads per row ----
  const int row = tid >> 4;
  const int cix = tid & 15;

  float m1 = -3.4e38f;
  for (int t = cix; t < 1024; t += 16) m1 = fmaxf(m1, sc[row * 1024 + t]);
  red[row * 32 + cix] = m1;
  __syncthreads();
  if (cix == 0) {
    float mm = red[row * 32];
    for (int i = 1; i < 16; ++i) mm = fmaxf(mm, red[row * 32 + i]);
    red[row * 32 + 16] = mm;
  }
  __syncthreads();
  m1 = red[row * 32 + 16];

  float s1 = 0.f;
  for (int t = cix; t < 1024; t += 16) s1 += expf(sc[row * 1024 + t] - m1);
  red[row * 32 + cix] = s1;
  __syncthreads();
  if (cix == 0) {
    float ss = 0.f;
    for (int i = 0; i < 16; ++i) ss += red[row * 32 + i];
    red[row * 32 + 17] = ss;
  }
  __syncthreads();
  const float inv1 = 1.0f / red[row * 32 + 17];

  float m2 = -3.4e38f;
  for (int t = cix; t < 1024; t += 16) {
    float p = expf(sc[row * 1024 + t] - m1) * inv1;
    float z = (float)mixs[row * 1024 + t] + (1.0f - ALPHA) * p;
    sc[row * 1024 + t] = z;
    m2 = fmaxf(m2, z);
  }
  red[row * 32 + cix] = m2;
  __syncthreads();
  if (cix == 0) {
    float mm = red[row * 32];
    for (int i = 1; i < 16; ++i) mm = fmaxf(mm, red[row * 32 + i]);
    red[row * 32 + 16] = mm;
  }
  __syncthreads();
  m2 = red[row * 32 + 16];

  float s2 = 0.f;
  for (int t = cix; t < 1024; t += 16) s2 += expf(sc[row * 1024 + t] - m2);
  red[row * 32 + cix] = s2;
  __syncthreads();
  if (cix == 0) {
    float ss = 0.f;
    for (int i = 0; i < 16; ++i) ss += red[row * 32 + i];
    red[row * 32 + 17] = ss;
  }
  __syncthreads();
  const float inv2 = 1.0f / red[row * 32 + 17];

  for (int t = cix; t < 1024; t += 16)
    pf[row * 1024 + t] = (_Float16)(expf(sc[row * 1024 + t] - m2) * inv2);
  __syncthreads();

  // ---- ctx = probs @ v : wave wv owns t-range [wv*128, wv*128+128) ----
  v8f accd[4] = {};
  for (int ks = 0; ks < 4; ++ks) {
    int tb = wv * 128 + ks * 32;
    v16h ap;
#pragma unroll
    for (int e = 0; e < 16; ++e)
      ap[e] = pf[mrowA * 1024 + tb + a16_k(lane, e)];
#pragma unroll
    for (int dt = 0; dt < 4; ++dt) {
      v16h bvf;
      int nc = lane & 15;
#pragma unroll
      for (int e = 0; e < 16; ++e)
        bvf[e] = vh[(size_t)(tb + a16_k(lane, e)) * HD + dt * 16 + nc];
      accd[dt] = __builtin_amdgcn_wmma_f32_16x16x32_f16(
          false, ap, false, bvf, (short)0, accd[dt], false, false);
    }
  }
  __syncthreads();

  // partials -> LDS (reuse sc), then cross-wave reduce + write out
#pragma unroll
  for (int dt = 0; dt < 4; ++dt)
#pragma unroll
    for (int r = 0; r < 8; ++r)
      sc[(wv * 4 + dt) * 256 + r * 32 + lane] = accd[dt][r];
  __syncthreads();

  for (int i = 0; i < 4; ++i) {
    int o   = tid + i * 256;      // 0..1023
    int dt  = o >> 8;
    int idx = o & 255;
    float sum = 0.f;
    for (int w2 = 0; w2 < 8; ++w2) sum += sc[(w2 * 4 + dt) * 256 + idx];
    int r = idx >> 5, l = idx & 31;
    int m = ((l >> 4) << 3) + r;
    int n = l & 15;
    out[((size_t)(b * SS + s0 + m)) * DD + h * HD + dt * 16 + n] = sum;
  }
}

// ---------------------------------------------------------------------------
extern "C" void kernel_launch(void* const* d_in, const int* in_sizes, int n_in,
                              void* d_out, int out_size, void* d_ws,
                              size_t ws_size, hipStream_t stream) {
  const float* hidden = (const float*)d_in[0];
  const float* mask   = (const float*)d_in[1];
  const float* prev   = (const float*)d_in[2];
  const float* Wq = (const float*)d_in[3];
  const float* bq = (const float*)d_in[4];
  const float* Wk = (const float*)d_in[5];
  const float* bk = (const float*)d_in[6];
  const float* Wv = (const float*)d_in[7];
  const float* bv = (const float*)d_in[8];
  const float* Wc = (const float*)d_in[9];
  const float* bc = (const float*)d_in[10];
  float* out = (float*)d_out;

  // workspace: q/k/v f16 (3 x 8 MB) + mixed_prev f16 (128 MB)
  const size_t QKV_ELEMS = (size_t)BB * HH * SS * HD;  // 4 Mi halves
  char* ws = (char*)d_ws;
  _Float16* qf = (_Float16*)ws;
  _Float16* kf = qf + QKV_ELEMS;
  _Float16* vf = kf + QKV_ELEMS;
  _Float16* mixed = (_Float16*)(ws + 3 * QKV_ELEMS * sizeof(_Float16));

  qkv_kernel<<<dim3(128, 16, 3), 256, 0, stream>>>(
      hidden, Wq, bq, Wk, bk, Wv, bv, qf, kf, vf);
  headmix_kernel<<<dim3(8, 1024, 4), 256, 0, stream>>>(prev, Wc, bc, mixed);
  attn_kernel<<<dim3(64, 16, 4), 256, 0, stream>>>(qf, kf, vf, mixed, mask,
                                                   out);
}
